// HypERPlus_17248588661540
// MI455X (gfx1250) — matrix operand order repaired and checked by reading, not compiled, not run on hardware
//
#include <hip/hip_runtime.h>
#include <hip/hip_bf16.h>
#include <math.h>

typedef __attribute__((ext_vector_type(8)))  _Float16 v8h;
typedef __attribute__((ext_vector_type(16))) _Float16 v16h;
typedef __attribute__((ext_vector_type(8)))  float    v8f;

#define B_SZ   4096
#define D1     200
#define W_IN   400      // 2*D1
#define OC     32
#define FW     9
#define W_OUT  392      // W_IN - FW + 1
#define FC_LEN 12544    // OC * W_OUT  (== 392 * 32, divisible by 32)
#define FC1_LEN 288
#define EPSF   1e-5f

// ---------------- P1: fc_w f32 -> f16 (row-major [400, 12544] preserved) ---
__global__ void cvt_fcw_kernel(const float* __restrict__ w,
                               _Float16* __restrict__ o, int n) {
  int i = blockIdx.x * blockDim.x + threadIdx.x;
  if (i < n) o[i] = (_Float16)w[i];
}

// ---------------- P2: hypernetwork filters k = r @ fc1_w^T + fc1_b ---------
__global__ __launch_bounds__(FC1_LEN)
void filters_kernel(const int* __restrict__ r_idx,
                    const float* __restrict__ R,
                    const float* __restrict__ w1,
                    const float* __restrict__ b1,
                    float* __restrict__ kf) {
  __shared__ float rs[D1];
  const int b = blockIdx.x;
  const int t = threadIdx.x;
  const float* rrow = R + (size_t)r_idx[b] * D1;
  for (int i = t; i < D1; i += FC1_LEN) rs[i] = rrow[i];
  __syncthreads();
  const float* wr = w1 + (size_t)t * D1;
  float s = 0.f;
  #pragma unroll 4
  for (int d = 0; d < D1; ++d) s += rs[d] * wr[d];
  kf[(size_t)b * FC1_LEN + t] = s + b1[t];
}

// ---------------- P3: gather + bn0 + per-example conv + bn1 -> flat (f16) --
__global__ __launch_bounds__(256)
void makeflat_kernel(const int* __restrict__ e1_idx,
                     const int* __restrict__ e2_idx,
                     const float* __restrict__ E,
                     const float* __restrict__ bn0g, const float* __restrict__ bn0b,
                     const float* __restrict__ bn0m, const float* __restrict__ bn0v,
                     const float* __restrict__ kf,
                     const float* __restrict__ g1, const float* __restrict__ be1,
                     const float* __restrict__ m1, const float* __restrict__ v1,
                     _Float16* __restrict__ flat) {
  __shared__ float xs[W_IN];
  __shared__ float ks[FC1_LEN];
  __shared__ float inv1s[OC], bet1s[OC];
  const int b = blockIdx.x, tid = threadIdx.x;

  const float s0 = bn0g[0] * rsqrtf(bn0v[0] + EPSF);
  const float o0 = bn0b[0] - bn0m[0] * s0;
  const float* e1 = E + (size_t)e1_idx[b] * D1;
  const float* e2 = E + (size_t)e2_idx[b] * D1;
  for (int i = tid; i < D1; i += 256) {
    xs[i]      = e1[i] * s0 + o0;
    xs[i + D1] = e2[i] * s0 + o0;
  }
  for (int i = tid; i < FC1_LEN; i += 256) ks[i] = kf[(size_t)b * FC1_LEN + i];
  if (tid < OC) {
    float iv = g1[tid] * rsqrtf(v1[tid] + EPSF);
    inv1s[tid] = iv;
    bet1s[tid] = be1[tid] - m1[tid] * iv;
  }
  __syncthreads();

  _Float16* fr = flat + (size_t)b * FC_LEN;
  for (int idx = tid; idx < FC_LEN; idx += 256) {
    const int o = idx / W_OUT;
    const int w = idx - o * W_OUT;
    const float* kk = &ks[o * FW];
    float s = 0.f;
    #pragma unroll
    for (int j = 0; j < FW; ++j) s += xs[w + j] * kk[j];
    s = s * inv1s[o] + bet1s[o];
    fr[idx] = (_Float16)s;   // coalesced f16 store
  }
}

// ---------------- P4: h = flat @ fc_w^T + fc_b, then bn2  (WMMA f16) -------
// Wave tile: 32 M x 80 N (2x5 accumulators = 80 VGPRs).
// B fragments (5 x v16h = 40 VGPRs) are loaded once per K-step and reused by
// both M-subtiles: 14 b128 loads per 10 WMMAs (~23 FLOP per L2 byte).
// Block = 4 waves along M -> 128 M rows. Grid: (4096/128, 400/80) = (32, 5).
__global__ __launch_bounds__(128)
void gemm_wmma_kernel(const _Float16* __restrict__ flat,
                      const _Float16* __restrict__ w16,
                      const float* __restrict__ fc_b,
                      const float* __restrict__ g2, const float* __restrict__ b2,
                      const float* __restrict__ m2, const float* __restrict__ v2,
                      float* __restrict__ h) {
  const int lane  = threadIdx.x & 31;
  const int wave  = threadIdx.x >> 5;
  const int lo16  = lane & 15;
  const int hi    = lane >> 4;           // 0: lanes 0-15, 1: lanes 16-31
  const int mBase = (blockIdx.x * 4 + wave) * 32;
  const int nBase = blockIdx.y * 80;

  v8f acc[2][5] = {};

  // A fragment (16-bit A 16x32 layout): lane holds row M=lo16;
  //   halves 0-7 = K (hi*8 ..), halves 8-15 = K (hi*8+16 ..)
  const _Float16* aRow0 = flat + (size_t)(mBase + lo16) * FC_LEN + hi * 8;
  const _Float16* aRow1 = aRow0 + (size_t)16 * FC_LEN;

  for (int kk = 0; kk < FC_LEN; kk += 32) {
    // B fragments (32x16): lane holds column N = nBase+t*16+lo16,
    //   halves 0-7 = K (hi*8 ..), halves 8-15 = K (hi*8+16 ..)
    v16h Bv[5];
    #pragma unroll
    for (int t = 0; t < 5; ++t) {
      const _Float16* bp =
          w16 + (size_t)(nBase + t * 16 + lo16) * FC_LEN + kk + hi * 8;
      v8h b0 = *(const v8h*)(bp);
      v8h b1 = *(const v8h*)(bp + 16);
      Bv[t] = __builtin_shufflevector(b0, b1,
          0,1,2,3,4,5,6,7,8,9,10,11,12,13,14,15);
    }

    v8h a0 = *(const v8h*)(aRow0 + kk);
    v8h a1 = *(const v8h*)(aRow0 + kk + 16);
    v16h A0 = __builtin_shufflevector(a0, a1,
        0,1,2,3,4,5,6,7,8,9,10,11,12,13,14,15);
    #pragma unroll
    for (int t = 0; t < 5; ++t)
      acc[0][t] = __builtin_amdgcn_wmma_f32_16x16x32_f16(
          false, A0, false, Bv[t], (short)0, acc[0][t], false, false);

    v8h a2 = *(const v8h*)(aRow1 + kk);
    v8h a3 = *(const v8h*)(aRow1 + kk + 16);
    v16h A1 = __builtin_shufflevector(a2, a3,
        0,1,2,3,4,5,6,7,8,9,10,11,12,13,14,15);
    #pragma unroll
    for (int t = 0; t < 5; ++t)
      acc[1][t] = __builtin_amdgcn_wmma_f32_16x16x32_f16(
          false, A1, false, Bv[t], (short)0, acc[1][t], false, false);
  }

  // Epilogue: + fc_b, then bn2.  C layout: VGPR i -> row base+i (+8 if hi).
  #pragma unroll
  for (int t = 0; t < 5; ++t) {
    const int col = nBase + t * 16 + lo16;
    const float inv = g2[col] * rsqrtf(v2[col] + EPSF);
    const float off = b2[col] - m2[col] * inv;
    const float fb  = fc_b[col];
    #pragma unroll
    for (int mt = 0; mt < 2; ++mt) {
      #pragma unroll
      for (int i = 0; i < 8; ++i) {
        const int row = mBase + mt * 16 + i + hi * 8;
        h[(size_t)row * W_IN + col] = (acc[mt][t][i] + fb) * inv + off;
      }
    }
  }
}

// ---------------- P5: out[b] = tanh(h[b,:] . fc2_w + fc2_b) + bias ---------
__global__ __launch_bounds__(256)
void out_kernel(const float* __restrict__ h,
                const float* __restrict__ w2,
                const float* __restrict__ fc2_b,
                const float* __restrict__ bias,
                float* __restrict__ out) {
  const int lane = threadIdx.x & 31;
  const int wave = threadIdx.x >> 5;
  const int row  = blockIdx.x * 8 + wave;
  const float* hr = h + (size_t)row * W_IN;
  float s = 0.f;
  for (int j = lane; j < W_IN; j += 32) s += hr[j] * w2[j];
  #pragma unroll
  for (int off = 16; off > 0; off >>= 1) s += __shfl_down(s, off, 32);
  if (lane == 0) out[row] = tanhf(s + fc2_b[0]) + bias[0];
}

// ---------------- launch ---------------------------------------------------
extern "C" void kernel_launch(void* const* d_in, const int* in_sizes, int n_in,
                              void* d_out, int out_size, void* d_ws, size_t ws_size,
                              hipStream_t stream) {
  (void)in_sizes; (void)n_in; (void)out_size; (void)ws_size;

  const int*   e1_idx = (const int*)  d_in[0];
  const int*   r_idx  = (const int*)  d_in[1];
  const int*   e2_idx = (const int*)  d_in[2];
  const float* E      = (const float*)d_in[3];
  const float* R      = (const float*)d_in[4];
  const float* bn0g = (const float*)d_in[5],  *bn0b = (const float*)d_in[6];
  const float* bn0m = (const float*)d_in[7],  *bn0v = (const float*)d_in[8];
  const float* fc1w = (const float*)d_in[9],  *fc1b = (const float*)d_in[10];
  const float* g1 = (const float*)d_in[11], *be1 = (const float*)d_in[12];
  const float* m1 = (const float*)d_in[13], *v1  = (const float*)d_in[14];
  const float* fcw = (const float*)d_in[15], *fcb = (const float*)d_in[16];
  const float* g2 = (const float*)d_in[17], *b2 = (const float*)d_in[18];
  const float* m2 = (const float*)d_in[19], *v2 = (const float*)d_in[20];
  const float* fc2w = (const float*)d_in[21], *fc2b = (const float*)d_in[22];
  const float* bias = (const float*)d_in[23];
  float* out = (float*)d_out;

  // Workspace layout (all 256B aligned):
  char* ws = (char*)d_ws;
  _Float16* w16  = (_Float16*)(ws);                         // 400*12544*2  = 10,035,200
  _Float16* flat = (_Float16*)(ws + 10035200);              // 4096*12544*2 = 102,760,448
  float*    kf   = (float*)   (ws + 10035200 + 102760448);  // 4096*288*4   = 4,718,592
  float*    hbuf = (float*)   (ws + 10035200 + 102760448 + 4718592); // 4096*400*4

  const int nW = W_IN * FC_LEN;  // 5,017,600
  cvt_fcw_kernel<<<(nW + 255) / 256, 256, 0, stream>>>(fcw, w16, nW);
  filters_kernel<<<B_SZ, FC1_LEN, 0, stream>>>(r_idx, R, fc1w, fc1b, kf);
  makeflat_kernel<<<B_SZ, 256, 0, stream>>>(e1_idx, e2_idx, E,
                                            bn0g, bn0b, bn0m, bn0v,
                                            kf, g1, be1, m1, v1, flat);
  gemm_wmma_kernel<<<dim3(B_SZ / 128, 5), 128, 0, stream>>>(
      flat, w16, fcb, g2, b2, m2, v2, hbuf);
  out_kernel<<<B_SZ / 8, 256, 0, stream>>>(hbuf, fc2w, fc2b, bias, out);
}
